// InnerAttention_40656160424181
// MI455X (gfx1250) — matrix operand order repaired and checked by reading, not compiled, not run on hardware
//
#include <hip/hip_runtime.h>
#include <hip/hip_bf16.h>

// ---------- types ----------
typedef _Float16 h8   __attribute__((ext_vector_type(8)));
typedef _Float16 v16h __attribute__((ext_vector_type(16)));
typedef float    v8f  __attribute__((ext_vector_type(8)));
typedef float    f4   __attribute__((ext_vector_type(4)));
typedef unsigned int u32x4 __attribute__((ext_vector_type(4)));
typedef int          i32x8 __attribute__((ext_vector_type(8)));
typedef int          i32x4 __attribute__((ext_vector_type(4)));

union F16x16 { v16h v; h8 h[2]; };

#define WMMA_F16(A, B, C) \
  __builtin_amdgcn_wmma_f32_16x16x32_f16(false, (A), false, (B), (short)0, (C), false, false)

// Tensor Data Mover availability (this toolchain: 6-arg clang-23 form)
#if defined(__has_builtin)
#  if __has_builtin(__builtin_amdgcn_tensor_load_to_lds)
#    define USE_TDM 1
#  endif
#endif
#ifndef USE_TDM
#  define USE_TDM 0
#endif

// Problem constants
#define NB   4
#define NH   16
#define NSEQ 2048
#define ND   64
#define BH   (NB*NH)

#if USE_TDM
// Issue a TDM load of a 64x64 f16 tile (row stride 64 elements) into LDS.
// D# per CDNA5 ISA: group0 = {count/flags, lds_addr,
// global_addr lo, global_addr hi | type=2}; group1 packs data_size=2B,
// tensor_dim0=64, tensor_dim1=2048, tile_dim0=64, tile_dim1=64, stride0=64.
__device__ __forceinline__ void tdm_load_tile64(const _Float16* gptr, unsigned lds_off) {
  unsigned long long ga = (unsigned long long)(uintptr_t)gptr;
  u32x4 g0;
  g0[0] = 1u;                                            // count=1, user mode
  g0[1] = lds_off;                                       // LDS byte address
  g0[2] = (unsigned)(ga & 0xFFFFFFFFu);                  // global_addr[31:0]
  g0[3] = (unsigned)((ga >> 32) & 0x1FFFFFFu) | (2u << 30); // addr[56:32] | type=2
  i32x8 g1;
  g1[0] = (int)(1u << 16);        // data_size = 1 (2 bytes); wg_mask = 0
  g1[1] = (int)(64u << 16);       // tensor_dim0[15:0] = 64
  g1[2] = (int)(2048u << 16);     // tensor_dim0 hi = 0 | tensor_dim1[15:0] = 2048
  g1[3] = (int)(64u << 16);       // tensor_dim1 hi = 0 | tile_dim0 = 64
  g1[4] = 64;                     // tile_dim1 = 64, tile_dim2 = 0
  g1[5] = 64;                     // tensor_dim0_stride = 64
  g1[6] = 0;
  g1[7] = 0;
  i32x4 z4 = {0, 0, 0, 0};        // 2D tensor: groups 2/3 unused
  i32x8 z8 = {0, 0, 0, 0, 0, 0, 0, 0};
  // 6-arg form: (group0, group1, group2, group3, group4/extra, cpol)
  __builtin_amdgcn_tensor_load_to_lds(g0, g1, z4, z4, z8, 0);
}
#endif

__device__ __forceinline__ unsigned lds_offset_of(const void* p) {
  // Flat LDS aperture address: low 32 bits are the LDS byte offset.
  return (unsigned)(uintptr_t)p;
}

// =====================================================================
// Kernel 1: QKV projection.  grid = BH*32 blocks of 128 threads.
// Each block: one (b,h) and one 64-row tile of N. Wt (f16, transposed W)
// staged in LDS, X rows as WMMA A fragments, 2x v_wmma per 16x16 tile.
// De-interleave stride-3 columns into q(*scale)/k/v f16 workspace.
// =====================================================================
__global__ __launch_bounds__(128) void qkv_proj_wmma(
    const float* __restrict__ x, const float* __restrict__ w,
    const float* __restrict__ bias,
    _Float16* __restrict__ qw, _Float16* __restrict__ kw, _Float16* __restrict__ vw)
{
  __shared__ _Float16 wt[192 * 64];   // wt[c*64 + d] = W[d][c]  (24 KB)

  const int bh   = blockIdx.x >> 5;
  const int tile = blockIdx.x & 31;
  const int h    = bh & (NH - 1);
  const int tid  = threadIdx.x;
  const int lane = tid & 31;
  const int wv   = tid >> 5;
  const int hw   = lane >> 4;        // half-wave id (0/1)
  const int ln   = lane & 15;

  // Stage W transposed as f16
  const float* wh = w + (size_t)h * 64 * 192;
  #pragma unroll 4
  for (int i = 0; i < 96; ++i) {
    int idx = i * 128 + tid;                 // coalesced over d*192+c
    int d = idx / 192, c = idx % 192;
    wt[c * 64 + d] = (_Float16)wh[idx];
  }
  __syncthreads();

  // A fragments: 16 rows of X, K = 0..63 (two 16x32 fragments), f32->f16
  const int r0  = tile * 64 + wv * 16;
  const int row = r0 + ln;
  const float* xrow = x + ((size_t)(bh * NSEQ + row)) * ND;
  F16x16 a[2];
  #pragma unroll
  for (int f = 0; f < 2; ++f) {
    f4 x0 = *(const f4*)(xrow + f * 32 + hw * 8);
    f4 x1 = *(const f4*)(xrow + f * 32 + hw * 8 + 4);
    f4 x2 = *(const f4*)(xrow + f * 32 + 16 + hw * 8);
    f4 x3 = *(const f4*)(xrow + f * 32 + 16 + hw * 8 + 4);
    h8 lo, hi;
    #pragma unroll
    for (int i = 0; i < 4; ++i) {
      lo[i] = (_Float16)x0[i]; lo[4 + i] = (_Float16)x1[i];
      hi[i] = (_Float16)x2[i]; hi[4 + i] = (_Float16)x3[i];
    }
    a[f].h[0] = lo; a[f].h[1] = hi;
  }

  const float* bb = bias + (size_t)h * 192;
  const int kb = hw * 8;
  #pragma unroll 2
  for (int nt = 0; nt < 12; ++nt) {
    const int c = nt * 16 + ln;              // output column 0..191
    F16x16 b0, b1;
    b0.h[0] = *(const h8*)&wt[c * 64 + kb];
    b0.h[1] = *(const h8*)&wt[c * 64 + kb + 16];
    b1.h[0] = *(const h8*)&wt[c * 64 + 32 + kb];
    b1.h[1] = *(const h8*)&wt[c * 64 + 32 + kb + 16];
    v8f acc = {};
    acc = WMMA_F16(a[0].v, b0.v, acc);
    acc = WMMA_F16(a[1].v, b1.v, acc);

    const float bv = bb[c];
    const int j = c % 3, d = c / 3;          // stride-3 de-interleave (q,k,v)
    _Float16* dst = (j == 0) ? qw : (j == 1) ? kw : vw;
    const float sc = (j == 0) ? 0.125f : 1.0f;  // q * D^-0.5
    #pragma unroll
    for (int r = 0; r < 8; ++r) {
      const int rr = r0 + r + (hw << 3);     // C layout: M = r + 8*halfwave
      dst[((size_t)(bh * NSEQ + rr)) * ND + d] = (_Float16)((acc[r] + bv) * sc);
    }
  }
}

// =====================================================================
// Kernel 2: flash attention.  grid = BH*32 blocks of 128 threads (4 waves).
// Each wave owns 16 query rows; loop over 32 key tiles of 64 rows.
// K/V tiles staged via TDM (tensor_load_to_lds + s_wait_tensorcnt),
// V transposed in LDS for the PV B-fragments, online softmax with
// width-16 shfl_xor reductions, P round-trip through per-wave LDS.
// =====================================================================
__global__ __launch_bounds__(128) void flash_attn_wmma(
    const _Float16* __restrict__ qw, const _Float16* __restrict__ kw,
    const _Float16* __restrict__ vw, float* __restrict__ out)
{
  __shared__ _Float16 lds_k [64 * 64];   // K tile,  row-major [key][d]
  __shared__ _Float16 lds_vr[64 * 64];   // V tile,  row-major [key][d]
  __shared__ _Float16 lds_vt[64 * 64];   // V tile,  transposed [d][key]
  __shared__ _Float16 lds_p [4 * 16 * 64]; // per-wave P tiles [qrow][key]

  const int bh  = blockIdx.x >> 5;
  const int qt  = blockIdx.x & 31;
  const int tid = threadIdx.x;
  const int lane = tid & 31;
  const int wv  = tid >> 5;
  const int hw  = lane >> 4;
  const int ln  = lane & 15;

  const size_t seqbase = (size_t)bh * NSEQ * ND;

  // Q fragments (already scaled by 0.125 in projection)
  const int qrow = qt * 64 + wv * 16 + ln;
  const _Float16* qr = qw + seqbase + (size_t)qrow * ND;
  F16x16 aq[2];
  #pragma unroll
  for (int f = 0; f < 2; ++f) {
    aq[f].h[0] = *(const h8*)(qr + f * 32 + hw * 8);
    aq[f].h[1] = *(const h8*)(qr + f * 32 + 16 + hw * 8);
  }

  v8f o[4] = {v8f{}, v8f{}, v8f{}, v8f{}};
  float m[8], l[8];
  #pragma unroll
  for (int r = 0; r < 8; ++r) { m[r] = -1e30f; l[r] = 0.0f; }

  for (int kt = 0; kt < 32; ++kt) {
    const _Float16* kg = kw + seqbase + (size_t)kt * 64 * ND;
    const _Float16* vg = vw + seqbase + (size_t)kt * 64 * ND;

#if USE_TDM
    if (tid < 32) {                       // wave 0 drives the TDM
      tdm_load_tile64(kg, lds_offset_of(lds_k));
      tdm_load_tile64(vg, lds_offset_of(lds_vr));
      __builtin_amdgcn_s_wait_tensorcnt(0);
    }
    __syncthreads();
#else
    #pragma unroll
    for (int g = 0; g < 4; ++g) {
      const int c = g * 128 + tid;        // 512 chunks of 8 halfs
      *(h8*)&lds_k [c * 8] = *(const h8*)(kg + c * 8);
      *(h8*)&lds_vr[c * 8] = *(const h8*)(vg + c * 8);
    }
    __syncthreads();
#endif

    // Transpose V in LDS: vt[d][key] = vr[key][d]
    #pragma unroll
    for (int g = 0; g < 4; ++g) {
      const int c = g * 128 + tid;
      const int key = c >> 3, d0 = (c & 7) * 8;
      h8 val = *(const h8*)&lds_vr[key * ND + d0];
      #pragma unroll
      for (int i = 0; i < 8; ++i) lds_vt[(d0 + i) * 64 + key] = val[i];
    }
    __syncthreads();

    // ---- S = Q * K^T  (4 tiles of 16 keys, K-dim = 64) ----
    v8f s[4];
    #pragma unroll
    for (int t = 0; t < 4; ++t) {
      const int key = t * 16 + ln;        // B column n = key; contiguous along d
      F16x16 b0, b1;
      b0.h[0] = *(const h8*)&lds_k[key * ND + hw * 8];
      b0.h[1] = *(const h8*)&lds_k[key * ND + 16 + hw * 8];
      b1.h[0] = *(const h8*)&lds_k[key * ND + 32 + hw * 8];
      b1.h[1] = *(const h8*)&lds_k[key * ND + 32 + 16 + hw * 8];
      v8f acc = {};
      acc = WMMA_F16(aq[0].v, b0.v, acc);
      acc = WMMA_F16(aq[1].v, b1.v, acc);
      s[t] = acc;
    }

    // ---- online softmax: row spans 16 lanes of a half-wave ----
    _Float16* pw = &lds_p[wv * 16 * 64];
    float alpha[8];
    #pragma unroll
    for (int r = 0; r < 8; ++r) {
      float mx = fmaxf(fmaxf(s[0][r], s[1][r]), fmaxf(s[2][r], s[3][r]));
      mx = fmaxf(mx, __shfl_xor(mx, 1, 16));
      mx = fmaxf(mx, __shfl_xor(mx, 2, 16));
      mx = fmaxf(mx, __shfl_xor(mx, 4, 16));
      mx = fmaxf(mx, __shfl_xor(mx, 8, 16));
      const float mn = fmaxf(m[r], mx);
      alpha[r] = __expf(m[r] - mn);
      m[r] = mn;
      float rs = 0.0f;
      #pragma unroll
      for (int t = 0; t < 4; ++t) {
        const float p = __expf(s[t][r] - mn);
        rs += p;
        pw[(r + hw * 8) * 64 + t * 16 + ln] = (_Float16)p;  // C->A via LDS
      }
      rs += __shfl_xor(rs, 1, 16);
      rs += __shfl_xor(rs, 2, 16);
      rs += __shfl_xor(rs, 4, 16);
      rs += __shfl_xor(rs, 8, 16);
      l[r] = l[r] * alpha[r] + rs;
    }
    #pragma unroll
    for (int nt = 0; nt < 4; ++nt)
      #pragma unroll
      for (int r = 0; r < 8; ++r) o[nt][r] *= alpha[r];

    // ---- P fragments from per-wave LDS (same-wave LDS is in-order) ----
    F16x16 ap[2];
    #pragma unroll
    for (int f = 0; f < 2; ++f) {
      ap[f].h[0] = *(const h8*)&pw[ln * 64 + f * 32 + hw * 8];
      ap[f].h[1] = *(const h8*)&pw[ln * 64 + f * 32 + 16 + hw * 8];
    }

    // ---- O += P * V  (4 d-tiles, K-dim = 64 keys) ----
    #pragma unroll
    for (int nt = 0; nt < 4; ++nt) {
      const int d = nt * 16 + ln;         // B column n = d; contiguous in vt
      F16x16 b0, b1;
      b0.h[0] = *(const h8*)&lds_vt[d * 64 + hw * 8];
      b0.h[1] = *(const h8*)&lds_vt[d * 64 + 16 + hw * 8];
      b1.h[0] = *(const h8*)&lds_vt[d * 64 + 32 + hw * 8];
      b1.h[1] = *(const h8*)&lds_vt[d * 64 + 32 + 16 + hw * 8];
      o[nt] = WMMA_F16(ap[0].v, b0.v, o[nt]);
      o[nt] = WMMA_F16(ap[1].v, b1.v, o[nt]);
    }
    __syncthreads();   // protect LDS tiles before next iteration's staging
  }

  // Epilogue: out = O / l   (f32)
  #pragma unroll
  for (int r = 0; r < 8; ++r) {
    const float inv = 1.0f / l[r];
    const int row = qt * 64 + wv * 16 + r + (hw << 3);
    float* orow = out + seqbase + (size_t)row * ND;
    orow[ 0 + ln] = o[0][r] * inv;
    orow[16 + ln] = o[1][r] * inv;
    orow[32 + ln] = o[2][r] * inv;
    orow[48 + ln] = o[3][r] * inv;
  }
}

// =====================================================================
extern "C" void kernel_launch(void* const* d_in, const int* in_sizes, int n_in,
                              void* d_out, int out_size, void* d_ws, size_t ws_size,
                              hipStream_t stream) {
  (void)in_sizes; (void)n_in; (void)out_size; (void)ws_size;
  const float* x    = (const float*)d_in[0];   // [B,H,N,D] f32
  const float* wqkv = (const float*)d_in[1];   // [H,D,3D] f32
  const float* bqkv = (const float*)d_in[2];   // [H,1,3D] f32
  float* out = (float*)d_out;                  // [B,H,N,D] f32

  _Float16* qws = (_Float16*)d_ws;             // [BH,N,D] each, f16
  const size_t per = (size_t)BH * NSEQ * ND;
  _Float16* kws = qws + per;
  _Float16* vws = kws + per;

  qkv_proj_wmma<<<BH * 32, 128, 0, stream>>>(x, wqkv, bqkv, qws, kws, vws);
  flash_attn_wmma<<<BH * 32, 128, 0, stream>>>(qws, kws, vws, out);
}
